// Pnetwork_76948634075831
// MI455X (gfx1250) — matrix operand order repaired and checked by reading, not compiled, not run on hardware
//
#include <hip/hip_runtime.h>
#include <hip/hip_bf16.h>

typedef __bf16 bf16_t;
typedef __attribute__((ext_vector_type(8)))  __bf16 v8bf;
typedef __attribute__((ext_vector_type(16))) __bf16 v16bf;
typedef __attribute__((ext_vector_type(8)))  float  v8f;
typedef __attribute__((ext_vector_type(4)))  unsigned v4u;
typedef __attribute__((ext_vector_type(8)))  int      v8i_t;
typedef __attribute__((ext_vector_type(4)))  int      v4i_t;

// ---- geometry ----
// rows = B*T*T = 128*16*16 = 32768, 128 rows per workgroup -> 256 WGs
// padded width: N = 112 (7 n-tiles), K = 128 (4 k-steps of 32)
#define MAT_ROWS   112
#define MAT_LDK    136                 // K stride (elems); 272B rows avoid bank conflicts
#define MAT_ELEMS  (MAT_ROWS * MAT_LDK)   // 15232
#define MAT_BYTES  (MAT_ELEMS * 2)        // 30464
#define N_MATS     101                    // W0, then Wa0,Wb0,...,Wa49,Wb49
#define XS_LD      120
#define AS_LD      136
#define XS_BYTES   (128 * XS_LD * 4)      // 61440
#define AS_BYTES   (128 * AS_LD * 2)      // 34816
#define WT_OFF     (XS_BYTES + AS_BYTES)  // 96256
#define SMEM_BYTES (WT_OFF + 2 * MAT_BYTES) // 157184
#define Y_OFF_BYTES ((size_t)N_MATS * MAT_BYTES) // y buffer after weight mats

// ---------------------------------------------------------------------------
// prep: transpose + pad fp32 weights -> bf16 (N rows x K cols), zero padding
// ---------------------------------------------------------------------------
__global__ void prep_weights(const float* __restrict__ W0,
                             const float* __restrict__ Wa,
                             const float* __restrict__ Wb,
                             bf16_t* __restrict__ mats) {
  int m = blockIdx.x;                       // 0..100
  bf16_t* dst = mats + (size_t)m * MAT_ELEMS;
  for (int e = threadIdx.x; e < MAT_ELEMS; e += 256) {
    int n = e / MAT_LDK;
    int k = e % MAT_LDK;
    float v = 0.f;
    if (m == 0) {
      if (k < 45 && n < 100) v = W0[k * 100 + n];
    } else {
      int b = (m - 1) >> 1;
      const float* W = ((m - 1) & 1) ? Wb : Wa;   // m odd -> Wa[b], m even -> Wb[b]
      if (k < 100 && n < 100) v = W[((size_t)b * 100 + k) * 100 + n];
    }
    dst[e] = (bf16_t)v;
  }
}

// ---------------------------------------------------------------------------
// TDM: async 1-D copy of one padded weight matrix (bf16) global -> LDS
// (6-arg builtin: g0, g1, g2, g3, g4, cpol)
// ---------------------------------------------------------------------------
__device__ __forceinline__ void tdm_load_mat(unsigned lds_off, const void* gsrc) {
  unsigned long long ga = (unsigned long long)gsrc;
  unsigned ne = (unsigned)MAT_ELEMS;        // 15232 2-byte elements (fits tile_dim0)
  v4u g0;
  g0[0] = 1u;                                // count=1, user descriptor
  g0[1] = lds_off;                           // lds_addr (bytes)
  g0[2] = (unsigned)ga;                      // global_addr[31:0]
  g0[3] = (unsigned)((ga >> 32) & 0x1FFFFFFull) | (2u << 30); // addr[56:32] | type=2
  v8i_t g1;
  g1[0] = (int)(1u << 16);                   // wg_mask=0, data_size=1 (2B)
  g1[1] = (int)((ne & 0xFFFFu) << 16);       // tensor_dim0[15:0] @ bits 63:48
  g1[2] = (int)((ne >> 16) & 0xFFFFu);       // tensor_dim0[31:16]
  g1[3] = (int)((ne & 0xFFFFu) << 16);       // tile_dim0 @ bits 127:112
  g1[4] = 0;                                 // tile_dim1 = tile_dim2 = 0
  g1[5] = (int)ne;                           // tensor_dim0_stride
  g1[6] = 0;
  g1[7] = 0;
  v4i_t z4 = {0, 0, 0, 0};
  v8i_t z8 = {0, 0, 0, 0, 0, 0, 0, 0};
  __builtin_amdgcn_tensor_load_to_lds(g0, g1, z4, z4, z8, 0);
}

__device__ __forceinline__ v16bf ld_frag(const bf16_t* p, int second_off) {
  v8bf lo = *(const v8bf*)p;
  v8bf hi = *(const v8bf*)(p + second_off);
  return __builtin_shufflevector(lo, hi, 0,1,2,3,4,5,6,7,8,9,10,11,12,13,14,15);
}

// ---------------------------------------------------------------------------
// net: fused gather + GEMM0 + 50 residual blocks + output head
// grid = 256 blocks x 256 threads (8 wave32), 128 rows per block
// ---------------------------------------------------------------------------
__global__ __launch_bounds__(256) void net_kernel(
    const float* __restrict__ inputs,
    const float* __restrict__ b1g, const float* __restrict__ b2g,
    const float* __restrict__ b3g, const float* __restrict__ b4g,
    const float* __restrict__ multg,
    const float* __restrict__ Woutg, const float* __restrict__ boutg,
    const bf16_t* __restrict__ mats, float* __restrict__ yg) {
  extern __shared__ __align__(16) char smem[];
  float*  xs  = (float*)smem;                          // 128 x 120 f32 residual
  bf16_t* As  = (bf16_t*)(smem + XS_BYTES);            // 128 x 136 bf16 operand
  bf16_t* Wts = (bf16_t*)(smem + WT_OFF);              // 2 weight slots

  const int tid  = threadIdx.x;
  const int lane = tid & 31;
  const int wv   = tid >> 5;
  const unsigned lds_base = (unsigned)(unsigned long long)(void*)smem;

  // ---- zero A operand buffer (padding must stay 0) ----
  {
    unsigned* az = (unsigned*)As;
    for (int e = tid; e < AS_BYTES / 4; e += 256) az[e] = 0u;
  }
  __syncthreads();

  // ---- gather 45 stencil features per row -> As (bf16) ----
  {
    int row  = tid & 127;
    int half = tid >> 7;
    size_t gr = (size_t)blockIdx.x * 128 + row;
    int bb = (int)(gr >> 8), ii = (int)((gr >> 4) & 15), jj = (int)(gr & 15);
    const float* inp = inputs + (size_t)bb * 9216;     // 32*32*9
    int ei = 2 * ii, oi2 = ei + 1, im = (ei + 31) & 31;
    int ej = 2 * jj, oj = ej + 1, jm = (ej + 31) & 31;
    const float* pat[5] = {
      inp + (oi2 * 32 + ej) * 9,   // right
      inp + (im  * 32 + ej) * 9,   // left
      inp + (ei  * 32 + oj) * 9,   // up
      inp + (ei  * 32 + jm) * 9,   // down
      inp + (ei  * 32 + ej) * 9 }; // center
    int p0 = half ? 3 : 0, p1 = half ? 5 : 3;
    for (int p = p0; p < p1; ++p)
      for (int q = 0; q < 9; ++q)
        As[row * AS_LD + p * 9 + q] = (bf16_t)pat[p][q];
  }

  // ---- prologue: TDM-load mats 0 and 1 into the two slots ----
  if (wv == 0) {
    tdm_load_mat(lds_base + WT_OFF,             mats);
    tdm_load_mat(lds_base + WT_OFF + MAT_BYTES, mats + MAT_ELEMS);
    __builtin_amdgcn_s_wait_tensorcnt(1);      // mat 0 resident
  }
  __syncthreads();

  const int nloc  = lane & 15;
  const int hi16  = (lane >> 4) & 1;
  const int mbase = (wv << 4) + (hi16 << 3);   // C-tile row base for this lane

  // ---- 101 chained GEMMs: g=0 -> W0; g=2b+1 -> Wa[b]; g=2b+2 -> Wb[b] ----
  for (int g = 0; g <= 100; ++g) {
    const bf16_t* W = Wts + (size_t)(g & 1) * MAT_ELEMS;
    v8f acc[7];
#pragma unroll
    for (int nt = 0; nt < 7; ++nt) acc[nt] = (v8f){0,0,0,0,0,0,0,0};

    const int ks = (g == 0) ? 2 : 4;
    for (int k = 0; k < ks; ++k) {
      // Batch all fragment loads for this k-step first so the ds_load_b128s
      // pipeline and the WMMA burst doesn't drain DScnt per instruction.
      const bf16_t* ap = As + (wv * 16 + nloc) * AS_LD + k * 32 + hi16 * 8;
      v16bf a = ld_frag(ap, 16);               // A 16x32: K 0-7/16-23 vs 8-15/24-31
      v16bf bm[7];
#pragma unroll
      for (int nt = 0; nt < 7; ++nt) {
        const bf16_t* bp = W + (nt * 16 + nloc) * MAT_LDK + k * 32 + hi16 * 8;
        bm[nt] = ld_frag(bp, 16);              // B 32x16: K 0-7/16-23 vs 8-15/24-31
      }
#pragma unroll
      for (int nt = 0; nt < 7; ++nt) {
        acc[nt] = __builtin_amdgcn_wmma_f32_16x16x32_bf16(
            false, a, false, bm[nt], (short)0, acc[nt], false, false);
      }
    }

    // ---- elementwise epilogue (own M-stripe only: no cross-wave hazard) ----
    float c0 = 0.f, c1 = 0.f, c2 = 1.f;        // per-variant scalars
    if (g == 0) {
      c0 = b1g[0];
    } else if (g & 1) {
      int bb = (g - 1) >> 1; c0 = b2g[bb]; c1 = b3g[bb];
    } else {
      int bb = (g - 2) >> 1; c0 = b4g[bb]; c2 = multg[bb];
      c1 = (bb + 1 < 50) ? b1g[bb + 1] : 0.f;
    }
#pragma unroll
    for (int nt = 0; nt < 7; ++nt) {
#pragma unroll
      for (int v = 0; v < 8; ++v) {
        int m = mbase + v, n = nt * 16 + nloc;
        float val = acc[nt][v];
        if (g == 0) {                           // x = relu(flat@W0)
          float x = fmaxf(val, 0.f);
          xs[m * XS_LD + n] = x;
          As[m * AS_LD + n] = (bf16_t)(x + c0); // + b1[0]
        } else if (g & 1) {                     // t = relu(2*x1+b2); A2 = 2t+b3
          float t = fmaxf(2.f * val + c0, 0.f);
          As[m * AS_LD + n] = (bf16_t)(2.f * t + c1);
        } else {                                // x = relu(x + x1*mult + b4)
          float xo = xs[m * XS_LD + n];
          float xn = fmaxf(xo + val * c2 + c0, 0.f);
          xs[m * XS_LD + n] = xn;
          As[m * AS_LD + n] = (bf16_t)(xn + c1); // + b1[b+1]
        }
      }
    }
    __syncthreads();                            // all waves done with slot g&1
    if (wv == 0) {
      if (g + 2 <= 100) {
        tdm_load_mat(lds_base + WT_OFF + (unsigned)(g & 1) * MAT_BYTES,
                     mats + (size_t)(g + 2) * MAT_ELEMS);
        __builtin_amdgcn_s_wait_tensorcnt(1);   // mat g+1 resident
      } else {
        __builtin_amdgcn_s_wait_tensorcnt(0);
      }
    }
    __syncthreads();                            // slot g+1 visible to all waves
  }

  // ---- output head: y = x @ Wout + bout (scalar, 2 outputs per thread) ----
  {
    int row = tid >> 1;
    int c   = (tid & 1) * 2;
    float d0 = boutg[c], d1 = boutg[c + 1];
    for (int k = 0; k < 100; ++k) {
      float xv = xs[row * XS_LD + k];
      d0 += xv * Woutg[k * 4 + c];
      d1 += xv * Woutg[k * 4 + c + 1];
    }
    size_t gr = (size_t)blockIdx.x * 128 + row;
    yg[gr * 4 + c]     = d0;
    yg[gr * 4 + c + 1] = d1;
  }
}

// ---------------------------------------------------------------------------
// post: contribution ratios + corner stencils -> (B,T,T,3,3)
// ---------------------------------------------------------------------------
__global__ void post_kernel(const float* __restrict__ inputs,
                            const float* __restrict__ yg,
                            float* __restrict__ out) {
  int r = blockIdx.x * 256 + threadIdx.x;       // 0..32767
  int b = r >> 8, i = (r >> 4) & 15, j = r & 15;
  const float* yb = yg + (size_t)b * 256 * 4;
  auto Y = [&](int ii, int jj, int c) { return yb[(ii * 16 + jj) * 4 + c]; };
  float y0 = Y(i, j, 0), y1 = Y(i, j, 1), y2 = Y(i, j, 2), y3 = Y(i, j, 3);
  float right_c = y0 / (Y((i + 1) & 15, j, 1) + y0);
  float left_c  = y1 / (y1 + Y((i + 15) & 15, j, 0));
  float up_c    = y2 / (y2 + Y(i, (j + 1) & 15, 3));
  float down_c  = y3 / (Y(i, (j + 15) & 15, 2) + y3);

  const float* inp = inputs + (size_t)b * 9216;
  int ei = 2 * i, oi2 = ei + 1, im = (ei + 31) & 31;
  int ej = 2 * j, oj = ej + 1, jm = (ej + 31) & 31;
  const float* oo  = inp + (oi2 * 32 + oj) * 9;
  const float* oid = inp + (oi2 * 32 + jm) * 9;
  const float* io  = inp + (im  * 32 + oj) * 9;
  const float* iid = inp + (im  * 32 + jm) * 9;
  float ru = -(oo[0]  + oo[3]  * right_c + oo[1]  * up_c)   / oo[4];
  float rd = -(oid[2] + oid[5] * right_c + oid[1] * down_c) / oid[4];
  float lu = -(io[6]  + io[3]  * left_c  + io[7]  * up_c)   / io[4];
  float ld = -(iid[8] + iid[5] * left_c  + iid[7] * down_c) / iid[4];

  float* o = out + (size_t)r * 9;
  o[0] = ld;     o[1] = left_c;  o[2] = lu;
  o[3] = down_c; o[4] = 1.f;     o[5] = up_c;
  o[6] = rd;     o[7] = right_c; o[8] = ru;
}

extern "C" void kernel_launch(void* const* d_in, const int* in_sizes, int n_in,
                              void* d_out, int out_size, void* d_ws, size_t ws_size,
                              hipStream_t stream) {
  (void)in_sizes; (void)n_in; (void)out_size; (void)ws_size;
  const float* inputs = (const float*)d_in[0];
  const float* W0     = (const float*)d_in[1];
  const float* Wa     = (const float*)d_in[2];
  const float* Wb     = (const float*)d_in[3];
  const float* b1     = (const float*)d_in[4];
  const float* b2     = (const float*)d_in[5];
  const float* b3     = (const float*)d_in[6];
  const float* b4     = (const float*)d_in[7];
  const float* mult   = (const float*)d_in[8];
  const float* Wout   = (const float*)d_in[9];
  const float* bout   = (const float*)d_in[10];

  bf16_t* mats = (bf16_t*)d_ws;
  float*  yg   = (float*)((char*)d_ws + Y_OFF_BYTES);
  float*  outp = (float*)d_out;

  prep_weights<<<N_MATS, 256, 0, stream>>>(W0, Wa, Wb, mats);
  net_kernel<<<256, 256, SMEM_BYTES, stream>>>(inputs, b1, b2, b3, b4, mult,
                                               Wout, bout, mats, yg);
  post_kernel<<<128, 256, 0, stream>>>(inputs, yg, outp);
}
